// PAM_and_CAM_39848706573446
// MI455X (gfx1250) — matrix operand order repaired and checked by reading, not compiled, not run on hardware
//
#include <hip/hip_runtime.h>
#include <stdint.h>

#define BN  16
#define CH  256
#define HH  64
#define WW  64
#define HWP 4096
#define PADW 66
#define K9  2304   // 9 * 256

typedef __attribute__((ext_vector_type(16))) __bf16 v16bf;
typedef __attribute__((ext_vector_type(8)))  float  v8f;

union FragU { v16bf v; uint4 q[2]; };

__device__ __forceinline__ unsigned short f2bf(float f) {
  unsigned int u = __float_as_uint(f);
  u += 0x7FFFu + ((u >> 16) & 1u);           // round-to-nearest-even
  return (unsigned short)(u >> 16);
}

// A fragment: 16x32 bf16, row-major weights; lane m = lane&15, hi = lane>>4.
// Per lane: K = [k0+8*hi .. +7] and [k0+16+8*hi .. +7]  (two 16B chunks)
__device__ __forceinline__ v16bf load_a_frag(const unsigned short* row_k0, int hi) {
  FragU f;
  f.q[0] = *(const uint4*)(row_k0 + 8 * hi);
  f.q[1] = *(const uint4*)(row_k0 + 16 + 8 * hi);
  return f.v;
}
// B fragment: 32x16 bf16; per lane (n fixed) K = [k0+8*hi .. +7] and [k0+16+8*hi .. +7]  (two 16B chunks)
__device__ __forceinline__ v16bf load_b_frag(const unsigned short* col_k0, int hi) {
  FragU f;
  f.q[0] = *(const uint4*)(col_k0 + 8 * hi);
  f.q[1] = *(const uint4*)(col_k0 + 16 + 8 * hi);
  return f.v;
}
__device__ __forceinline__ v8f wmma_bf16(v16bf a, v16bf b, v8f c) {
  return __builtin_amdgcn_wmma_f32_16x16x32_bf16(false, a, false, b, (short)0, c, false, false);
}

// 64x64 D tile staged transposed in LDS (rows = n_local, cols = m_local, stride 72),
// flushed as coalesced rows of 64 bf16 (dst row stride = 256 elements).
__device__ __forceinline__ void flush_tileT(unsigned short* dst, const unsigned short* lds) {
  int t = threadIdx.x;
  int row = t >> 1, half = t & 1;
  const uint4* s = (const uint4*)(lds + row * 72 + half * 32);
  uint4* d = (uint4*)(dst + (size_t)row * CH + half * 32);
  d[0] = s[0]; d[1] = s[1]; d[2] = s[2]; d[3] = s[3];
}

// ---------------- small / prep kernels ----------------

__global__ void k_fill0(uint4* p, size_t n) {
  size_t i = (size_t)blockIdx.x * blockDim.x + threadIdx.x;
  size_t stride = (size_t)gridDim.x * blockDim.x;
  uint4 z; z.x = z.y = z.z = z.w = 0u;
  for (; i < n; i += stride) p[i] = z;
}

// convert pa_v_w/ca_v_w to bf16, reorder conv weights to [o][tap][c] bf16
__global__ void k_weights(const float* pvw, const float* cvw,
                          const float* cpw, const float* ccw,
                          unsigned short* WvPA, unsigned short* WvCA,
                          unsigned short* WpaR, unsigned short* WcaR) {
  int i = blockIdx.x * blockDim.x + threadIdx.x;
  int stride = gridDim.x * blockDim.x;
  const int n1 = CH * CH;        // 65536
  const int n2 = CH * K9;        // 589824
  const int total = 2 * n1 + 2 * n2;
  for (; i < total; i += stride) {
    if (i < n1) {
      WvPA[i] = f2bf(pvw[i]);
    } else if (i < 2 * n1) {
      WvCA[i - n1] = f2bf(cvw[i - n1]);
    } else if (i < 2 * n1 + n2) {
      int j = i - 2 * n1;
      int o = j / K9, r = j % K9, t = r / CH, c = r % CH;
      WpaR[j] = f2bf(cpw[(o * CH + c) * 9 + t]);
    } else {
      int j = i - 2 * n1 - n2;
      int o = j / K9, r = j % K9, t = r / CH, c = r % CH;
      WcaR[j] = f2bf(ccw[(o * CH + c) * 9 + t]);
    }
  }
}

// xT[b][n][c] bf16 from x[b][c][n] f32 (LDS tiled transpose)
__global__ void k_transpose(const float* x, unsigned short* xT) {
  __shared__ float tile[32][33];
  int b = blockIdx.z;
  int n0 = blockIdx.x * 32, c0 = blockIdx.y * 32;
  int tx = threadIdx.x, ty = threadIdx.y;
  for (int j = 0; j < 4; j++) {
    int c = c0 + ty + 8 * j;
    tile[ty + 8 * j][tx] = x[((size_t)(b * CH + c)) * HWP + n0 + tx];
  }
  __syncthreads();
  for (int j = 0; j < 4; j++) {
    int n = n0 + ty + 8 * j;
    xT[((size_t)(b * HWP + n)) * CH + c0 + tx] = f2bf(tile[tx][ty + 8 * j]);
  }
}

// per (b,c): row means (over W), col means (over H), global mean
__global__ void k_reduce(const float* x, float* xmW, float* xmH, float* xm) {
  int bc = blockIdx.x, t = threadIdx.x;
  const float* p = x + (size_t)bc * HWP;
  float rs = 0.f;
  for (int w = 0; w < WW; w++) rs += p[t * WW + w];
  xmW[bc * HH + t] = rs * (1.f / WW);
  float cs = 0.f;
  for (int h = 0; h < HH; h++) cs += p[h * WW + t];
  xmH[bc * WW + t] = cs * (1.f / HH);
  __shared__ float red[64];
  red[t] = rs; __syncthreads();
  for (int s = 32; s > 0; s >>= 1) { if (t < s) red[t] += red[t + s]; __syncthreads(); }
  if (t == 0) xm[bc] = red[0] * (1.f / (HH * WW));
}

// tiny conv1x1 GEMMs on the reduced tensors (mean commutes with 1x1 conv)
__global__ void k_small(const float* qw, const float* qb, const float* kw, const float* kb,
                        const float* cqw, const float* cqb, const float* ckw, const float* ckb,
                        const float* xmW, const float* xmH, const float* xm,
                        float* paq, float* pak, float* caq, float* cak) {
  int bo = blockIdx.x;
  int b = bo >> 8, o = bo & 255, t = threadIdx.x;
  float sq = 0.f, sk = 0.f;
  for (int c = 0; c < CH; c++) {
    sq += qw[o * CH + c] * xmW[(b * CH + c) * HH + t];
    sk += kw[o * CH + c] * xmH[(b * CH + c) * WW + t];
  }
  paq[bo * HH + t] = sq + qb[o];
  pak[bo * WW + t] = sk + kb[o];
  float cq = 0.f, ck = 0.f;
  for (int c = t; c < CH; c += 64) {
    float xv = xm[b * CH + c];
    cq += cqw[o * CH + c] * xv;
    ck += ckw[o * CH + c] * xv;
  }
  __shared__ float r1[64], r2[64];
  r1[t] = cq; r2[t] = ck; __syncthreads();
  for (int s = 32; s > 0; s >>= 1) {
    if (t < s) { r1[t] += r1[t + s]; r2[t] += r2[t + s]; }
    __syncthreads();
  }
  if (t == 0) { caq[bo] = r1[0] + cqb[o]; cak[bo] = r2[0] + ckb[o]; }
}

// per (b,c): softmax stats of rank-1 energy q_h * k_w over the 64x64 map
__global__ void k_pam_stats(const float* paq, const float* pak, float* pm, float* piz) {
  int bo = blockIdx.x, t = threadIdx.x;
  __shared__ float qs[64], ks[64], red[64], msh;
  qs[t] = paq[bo * 64 + t];
  ks[t] = pak[bo * 64 + t];
  __syncthreads();
  float q = qs[t];
  float mx = -3.4e38f;
  for (int w = 0; w < 64; w++) mx = fmaxf(mx, q * ks[w]);
  red[t] = mx; __syncthreads();
  for (int s = 32; s > 0; s >>= 1) { if (t < s) red[t] = fmaxf(red[t], red[t + s]); __syncthreads(); }
  if (t == 0) msh = red[0];
  __syncthreads();
  float m = msh;
  float se = 0.f;
  for (int w = 0; w < 64; w++) se += __expf(q * ks[w] - m);
  red[t] = se; __syncthreads();
  for (int s = 32; s > 0; s >>= 1) { if (t < s) red[t] += red[t + s]; __syncthreads(); }
  if (t == 0) { pm[bo] = m; piz[bo] = 1.f / red[0]; }
}

// per batch: softmax over the full C*C energy map, emit bf16 attention matrix
__global__ void k_cam_att(const float* caq, const float* cak, unsigned short* attB) {
  int b = blockIdx.x, t = threadIdx.x;
  __shared__ float qs[256], ks[256], red[256], sm;
  qs[t] = caq[b * 256 + t];
  ks[t] = cak[b * 256 + t];
  __syncthreads();
  float k = ks[t];
  float mx = -3.4e38f;
  for (int i = 0; i < 256; i++) mx = fmaxf(mx, qs[i] * k);
  red[t] = mx; __syncthreads();
  for (int s = 128; s > 0; s >>= 1) { if (t < s) red[t] = fmaxf(red[t], red[t + s]); __syncthreads(); }
  if (t == 0) sm = red[0];
  __syncthreads();
  float m = sm;
  float se = 0.f;
  for (int i = 0; i < 256; i++) se += __expf(qs[i] * k - m);
  red[t] = se; __syncthreads();
  for (int s = 128; s > 0; s >>= 1) { if (t < s) red[t] += red[t + s]; __syncthreads(); }
  if (t == 0) sm = red[0];
  __syncthreads();
  float iz = 1.f / sm;
  for (int i = 0; i < 256; i++)
    attB[(size_t)b * 65536 + i * 256 + t] = f2bf(__expf(qs[i] * k - m) * iz);
}

// ---------------- WMMA GEMM kernels ----------------
// block = 128 threads (4 waves); block tile 64(M) x 64(N); wave = 16(M) x 64(N)

// pam = gamma * ((Wv @ x + vb) * softmax(q k^T)) + x  -> padded transposed bf16
__global__ __launch_bounds__(128)
void k_gemm_pam(const unsigned short* xT, const unsigned short* Wv, const float* vb,
                const float* paq, const float* pak, const float* pm, const float* piz,
                const float* x, const float* gamma, unsigned short* pamP) {
  int b = blockIdx.z, m0 = blockIdx.y * 64, y = blockIdx.x;
  int tid = threadIdx.x, wv = tid >> 5, lane = tid & 31, hi = lane >> 4, ln = lane & 15;
  const unsigned short* Abase = Wv + (size_t)(m0 + wv * 16 + ln) * CH;
  const unsigned short* Bbase = xT + ((size_t)(b * HWP + y * 64 + ln)) * CH;
  v8f acc[4];
  for (int s = 0; s < 4; s++)
    for (int r = 0; r < 8; r++) acc[s][r] = 0.f;
  for (int k0 = 0; k0 < CH; k0 += 32) {
    v16bf af = load_a_frag(Abase + k0, hi);
#pragma unroll
    for (int s = 0; s < 4; s++) {
      v16bf bf = load_b_frag(Bbase + (size_t)s * 16 * CH + k0, hi);
      acc[s] = wmma_bf16(af, bf, acc[s]);
    }
  }
  __shared__ __attribute__((aligned(16))) unsigned short lds[64 * 72];
  float g = gamma[0];
#pragma unroll
  for (int s = 0; s < 4; s++) {
    int w = s * 16 + ln;
#pragma unroll
    for (int r = 0; r < 8; r++) {
      int o = m0 + wv * 16 + r + 8 * hi;
      int bo = b * CH + o;
      float att = __expf(paq[bo * HH + y] * pak[bo * WW + w] - pm[bo]) * piz[bo];
      float val = acc[s][r] + vb[o];
      float xv = x[(size_t)bo * HWP + y * 64 + w];
      lds[w * 72 + (wv * 16 + r + 8 * hi)] = f2bf(g * (val * att) + xv);
    }
  }
  __syncthreads();
  flush_tileT(pamP + (((size_t)b * PADW + (y + 1)) * PADW + 1) * CH + m0, lds);
}

// v2 = ca_v_w @ x + cvb  -> transposed bf16 [b][n][j]
__global__ __launch_bounds__(128)
void k_gemm_v2(const unsigned short* xT, const unsigned short* Wv, const float* vb,
               unsigned short* v2T) {
  int b = blockIdx.z, m0 = blockIdx.y * 64, nt = blockIdx.x;
  int tid = threadIdx.x, wv = tid >> 5, lane = tid & 31, hi = lane >> 4, ln = lane & 15;
  const unsigned short* Abase = Wv + (size_t)(m0 + wv * 16 + ln) * CH;
  const unsigned short* Bbase = xT + ((size_t)(b * HWP + nt * 64 + ln)) * CH;
  v8f acc[4];
  for (int s = 0; s < 4; s++)
    for (int r = 0; r < 8; r++) acc[s][r] = 0.f;
  for (int k0 = 0; k0 < CH; k0 += 32) {
    v16bf af = load_a_frag(Abase + k0, hi);
#pragma unroll
    for (int s = 0; s < 4; s++) {
      v16bf bf = load_b_frag(Bbase + (size_t)s * 16 * CH + k0, hi);
      acc[s] = wmma_bf16(af, bf, acc[s]);
    }
  }
  __shared__ __attribute__((aligned(16))) unsigned short lds[64 * 72];
#pragma unroll
  for (int s = 0; s < 4; s++) {
    int nl = s * 16 + ln;
#pragma unroll
    for (int r = 0; r < 8; r++) {
      int o = m0 + wv * 16 + r + 8 * hi;
      lds[nl * 72 + (wv * 16 + r + 8 * hi)] = f2bf(acc[s][r] + vb[o]);
    }
  }
  __syncthreads();
  flush_tileT(v2T + ((size_t)(b * HWP + nt * 64)) * CH + m0, lds);
}

// cam = gamma * (att @ v2) + x  -> padded transposed bf16
__global__ __launch_bounds__(128)
void k_gemm_cam(const unsigned short* attB, const unsigned short* v2T,
                const float* x, const float* gamma, unsigned short* camP) {
  int b = blockIdx.z, m0 = blockIdx.y * 64, y = blockIdx.x;
  int tid = threadIdx.x, wv = tid >> 5, lane = tid & 31, hi = lane >> 4, ln = lane & 15;
  const unsigned short* Abase = attB + (size_t)b * CH * CH + (size_t)(m0 + wv * 16 + ln) * CH;
  const unsigned short* Bbase = v2T + ((size_t)(b * HWP + y * 64 + ln)) * CH;
  v8f acc[4];
  for (int s = 0; s < 4; s++)
    for (int r = 0; r < 8; r++) acc[s][r] = 0.f;
  for (int k0 = 0; k0 < CH; k0 += 32) {
    v16bf af = load_a_frag(Abase + k0, hi);
#pragma unroll
    for (int s = 0; s < 4; s++) {
      v16bf bf = load_b_frag(Bbase + (size_t)s * 16 * CH + k0, hi);
      acc[s] = wmma_bf16(af, bf, acc[s]);
    }
  }
  __shared__ __attribute__((aligned(16))) unsigned short lds[64 * 72];
  float g = gamma[0];
#pragma unroll
  for (int s = 0; s < 4; s++) {
    int w = s * 16 + ln;
#pragma unroll
    for (int r = 0; r < 8; r++) {
      int o = m0 + wv * 16 + r + 8 * hi;
      float xv = x[(size_t)(b * CH + o) * HWP + y * 64 + w];
      lds[w * 72 + (wv * 16 + r + 8 * hi)] = f2bf(g * acc[s][r] + xv);
    }
  }
  __syncthreads();
  flush_tileT(camP + (((size_t)b * PADW + (y + 1)) * PADW + 1) * CH + m0, lds);
}

// fused dual 3x3 conv (implicit GEMM over zero-padded transposed inputs), f32 out
__global__ __launch_bounds__(128)
void k_conv(const unsigned short* pamP, const unsigned short* camP,
            const unsigned short* WpaR, const unsigned short* WcaR,
            const float* bpa, const float* bca, float* out) {
  int b = blockIdx.z, m0 = blockIdx.y * 64, y = blockIdx.x;
  int tid = threadIdx.x, wv = tid >> 5, lane = tid & 31, hi = lane >> 4, ln = lane & 15;
  v8f acc[4];
  for (int s = 0; s < 4; s++)
    for (int r = 0; r < 8; r++) acc[s][r] = 0.f;
  for (int src = 0; src < 2; src++) {
    const unsigned short* Ab = (src == 0 ? WpaR : WcaR) + (size_t)(m0 + wv * 16 + ln) * K9;
    const unsigned short* P  = (src == 0 ? pamP : camP) + (size_t)b * PADW * PADW * CH;
    for (int t = 0; t < 9; t++) {
      int dy = t / 3, dx = t % 3;
      const unsigned short* Brow = P + ((size_t)(y + dy) * PADW) * CH;
      for (int k0 = 0; k0 < CH; k0 += 32) {
        v16bf af = load_a_frag(Ab + t * CH + k0, hi);
#pragma unroll
        for (int s = 0; s < 4; s++) {
          int w = s * 16 + ln;
          acc[s] = wmma_bf16(af, load_b_frag(Brow + (size_t)(w + dx) * CH + k0, hi), acc[s]);
        }
      }
    }
  }
#pragma unroll
  for (int s = 0; s < 4; s++) {
    int w = s * 16 + ln;
#pragma unroll
    for (int r = 0; r < 8; r++) {
      int o = m0 + wv * 16 + r + 8 * hi;
      out[((size_t)(b * CH + o)) * HWP + y * 64 + w] = acc[s][r] + bpa[o] + bca[o];
    }
  }
}

// ---------------- host launcher ----------------

extern "C" void kernel_launch(void* const* d_in, const int* in_sizes, int n_in,
                              void* d_out, int out_size, void* d_ws, size_t ws_size,
                              hipStream_t stream) {
  (void)in_sizes; (void)n_in; (void)out_size; (void)ws_size;
  const float* x         = (const float*)d_in[0];
  const float* pa_q_w    = (const float*)d_in[1];
  const float* pa_q_b    = (const float*)d_in[2];
  const float* pa_k_w    = (const float*)d_in[3];
  const float* pa_k_b    = (const float*)d_in[4];
  const float* pa_v_w    = (const float*)d_in[5];
  const float* pa_v_b    = (const float*)d_in[6];
  const float* pa_gamma  = (const float*)d_in[7];
  const float* ca_q_w    = (const float*)d_in[8];
  const float* ca_q_b    = (const float*)d_in[9];
  const float* ca_k_w    = (const float*)d_in[10];
  const float* ca_k_b    = (const float*)d_in[11];
  const float* ca_v_w    = (const float*)d_in[12];
  const float* ca_v_b    = (const float*)d_in[13];
  const float* ca_gamma  = (const float*)d_in[14];
  const float* conv_pa_w = (const float*)d_in[15];
  const float* conv_pa_b = (const float*)d_in[16];
  const float* conv_ca_w = (const float*)d_in[17];
  const float* conv_ca_b = (const float*)d_in[18];

  char* ws = (char*)d_ws;
  size_t off = 0;
  auto alloc = [&](size_t bytes) -> char* {
    char* p = ws + off;
    off = (off + bytes + 255) & ~(size_t)255;
    return p;
  };
  unsigned short* xT   = (unsigned short*)alloc((size_t)BN * HWP * CH * 2);
  unsigned short* pamP = (unsigned short*)alloc((size_t)BN * PADW * PADW * CH * 2);
  unsigned short* camP = (unsigned short*)alloc((size_t)BN * PADW * PADW * CH * 2); // contiguous after pamP
  unsigned short* v2T  = (unsigned short*)alloc((size_t)BN * HWP * CH * 2);
  unsigned short* attB = (unsigned short*)alloc((size_t)BN * CH * CH * 2);
  unsigned short* WvPA = (unsigned short*)alloc((size_t)CH * CH * 2);
  unsigned short* WvCA = (unsigned short*)alloc((size_t)CH * CH * 2);
  unsigned short* WpaR = (unsigned short*)alloc((size_t)CH * K9 * 2);
  unsigned short* WcaR = (unsigned short*)alloc((size_t)CH * K9 * 2);
  float* xmW = (float*)alloc((size_t)BN * CH * HH * 4);
  float* xmH = (float*)alloc((size_t)BN * CH * WW * 4);
  float* xm  = (float*)alloc((size_t)BN * CH * 4);
  float* paq = (float*)alloc((size_t)BN * CH * HH * 4);
  float* pak = (float*)alloc((size_t)BN * CH * WW * 4);
  float* caq = (float*)alloc((size_t)BN * CH * 4);
  float* cak = (float*)alloc((size_t)BN * CH * 4);
  float* pm  = (float*)alloc((size_t)BN * CH * 4);
  float* piz = (float*)alloc((size_t)BN * CH * 4);

  // zero both padded buffers (pamP and camP are contiguous, sizes 256B-aligned)
  size_t fillN = ((size_t)2 * BN * PADW * PADW * CH * 2) / 16;
  k_fill0<<<2048, 256, 0, stream>>>((uint4*)pamP, fillN);

  k_weights<<<2560, 256, 0, stream>>>(pa_v_w, ca_v_w, conv_pa_w, conv_ca_w,
                                      WvPA, WvCA, WpaR, WcaR);
  k_transpose<<<dim3(HWP / 32, CH / 32, BN), dim3(32, 8), 0, stream>>>(x, xT);
  k_reduce<<<BN * CH, 64, 0, stream>>>(x, xmW, xmH, xm);
  k_small<<<BN * CH, 64, 0, stream>>>(pa_q_w, pa_q_b, pa_k_w, pa_k_b,
                                      ca_q_w, ca_q_b, ca_k_w, ca_k_b,
                                      xmW, xmH, xm, paq, pak, caq, cak);
  k_pam_stats<<<BN * CH, 64, 0, stream>>>(paq, pak, pm, piz);
  k_cam_att<<<BN, 256, 0, stream>>>(caq, cak, attB);

  dim3 gg(HWP / 64, CH / 64, BN);
  k_gemm_pam<<<gg, 128, 0, stream>>>(xT, WvPA, pa_v_b, paq, pak, pm, piz, x, pa_gamma, pamP);
  k_gemm_v2<<<gg, 128, 0, stream>>>(xT, WvCA, ca_v_b, v2T);
  k_gemm_cam<<<gg, 128, 0, stream>>>(attB, v2T, x, ca_gamma, camP);
  k_conv<<<gg, 128, 0, stream>>>(pamP, camP, WpaR, WcaR, conv_pa_b, conv_ca_b, (float*)d_out);
}
